// EncoderBlock_53266184405825
// MI455X (gfx1250) — matrix operand order repaired and checked by reading, not compiled, not run on hardware
//
#include <hip/hip_runtime.h>
#include <hip/hip_bf16.h>
#include <math.h>

typedef __attribute__((ext_vector_type(16))) _Float16 v16h;
typedef __attribute__((ext_vector_type(8)))  _Float16 v8h;
typedef __attribute__((ext_vector_type(8)))  float    v8f;

union FragH { v16h f; v8h h[2]; _Float16 e[16]; };

__device__ __forceinline__ v8f wmma16(const FragH& a, const FragH& b, v8f c) {
  return __builtin_amdgcn_wmma_f32_16x16x32_f16(false, a.f, false, b.f, (short)0, c, false, false);
}

__device__ __forceinline__ void frag_zero(FragH& z) {
#pragma unroll
  for (int i = 0; i < 16; ++i) z.e[i] = (_Float16)0.f;
}

// ---------------- elementwise helpers ----------------

__global__ __launch_bounds__(256) void k_transpose_in(const float* __restrict__ x,
                                                      float* __restrict__ act,
                                                      int C, int T, int total) {
  int i = blockIdx.x * 256 + threadIdx.x;
  if (i >= total) return;
  int t = i % T;
  int rem = i / T;
  int c = rem % C;
  int b = rem / C;
  act[((size_t)(b * T + t)) * C + c] = x[i];
}

__global__ __launch_bounds__(256) void k_convert_w(const float* __restrict__ w,
                                                   _Float16* __restrict__ w16,
                                                   int O, int I, int K, int total) {
  int idx = blockIdx.x * 256 + threadIdx.x;
  if (idx >= total) return;
  int i = idx % I;
  int rem = idx / I;
  int o = rem % O;
  int k = rem / O;
  w16[((size_t)k * O + o) * I + i] = (_Float16)w[((size_t)o * I + i) * K + k];
}

__global__ __launch_bounds__(256) void k_elu16(const float* __restrict__ in,
                                               _Float16* __restrict__ out, int n) {
  int i = blockIdx.x * 256 + threadIdx.x;
  if (i >= n) return;
  float v = in[i];
  out[i] = (_Float16)(v > 0.f ? v : expm1f(v));
}

__global__ __launch_bounds__(256) void k_vtrans(const _Float16* __restrict__ qkv,
                                                _Float16* __restrict__ vt) {
  int idx = blockIdx.x * 256 + threadIdx.x;   // total = B*H*64*2048
  int t = idx & 2047;
  int rem = idx >> 11;
  int d = rem & 63;
  rem >>= 6;
  int h = rem & 7;
  int b = rem >> 3;
  vt[idx] = qkv[((size_t)(b * 2048 + t)) * 1536 + 1024 + h * 64 + d];
}

// ---------------- WMMA GEMM epilogue ----------------
// modes: 0 = f16, 1 = f16 GELU(exact), 2 = f32 +=, 3 = f32 =, 4 = f16 ELU
__device__ __forceinline__ void store_tile(v8f c, int o_base, size_t n_base, int lm, int lh,
                                           int Mout, const float* bias,
                                           float* fout, _Float16* hout, int mode) {
  size_t n = n_base + lm;
#pragma unroll
  for (int r = 0; r < 8; ++r) {
    int o = o_base + r + 8 * lh;
    float v = c[r] + (bias ? bias[o] : 0.f);
    size_t idx = n * (size_t)Mout + o;
    if (mode == 0) {
      hout[idx] = (_Float16)v;
    } else if (mode == 1) {
      v = 0.5f * v * (1.f + erff(v * 0.70710678118654752f));
      hout[idx] = (_Float16)v;
    } else if (mode == 2) {
      fout[idx] += v;
    } else if (mode == 3) {
      fout[idx] = v;
    } else {
      v = v > 0.f ? v : expm1f(v);
      hout[idx] = (_Float16)v;
    }
  }
}

// ---------------- generic WMMA linear: out[n, o] = act(bias[o] + sum_k w[o,k] in[n,k]) --------
// grid.x = Mout/128, grid.y = NT/256; block = 256 (8 waves as 2x4 -> 128x256 block tile;
// wave tile = 64x64: 16 accumulators, 8 fragment loads, 16 WMMAs per K-step -> 32 FLOP/B)
__global__ __launch_bounds__(256) void k_wmma_linear(const _Float16* __restrict__ in16,
                                                     const _Float16* __restrict__ w16,
                                                     const float* __restrict__ bias,
                                                     float* __restrict__ fout,
                                                     _Float16* __restrict__ hout,
                                                     int Kin, int Mout, int mode) {
  const int lane = threadIdx.x & 31, wave = threadIdx.x >> 5;
  const int lm = lane & 15, lh = lane >> 4;
  const int mbase = blockIdx.x * 128 + (wave & 1) * 64;
  const size_t nbase = (size_t)blockIdx.y * 256 + (wave >> 1) * 64;

  const _Float16* ap[4];
  const _Float16* bp[4];
#pragma unroll
  for (int i = 0; i < 4; ++i) {
    ap[i] = w16 + (size_t)(mbase + 16 * i + lm) * Kin + lh * 16;
    bp[i] = in16 + (nbase + 16 * i + lm) * (size_t)Kin + lh * 16;
  }

  v8f acc[4][4] = {};
  for (int k0 = 0; k0 < Kin; k0 += 32) {
    FragH a[4], b[4];
#pragma unroll
    for (int i = 0; i < 4; ++i) {
      a[i].h[0] = *(const v8h*)(ap[i] + k0);
      a[i].h[1] = *(const v8h*)(ap[i] + k0 + 8);
      b[i].f = *(const v16h*)(bp[i] + k0);
    }
    __builtin_prefetch(ap[0] + k0 + 128, 0, 1);
#pragma unroll
    for (int i = 0; i < 4; ++i)
#pragma unroll
      for (int j = 0; j < 4; ++j)
        acc[i][j] = wmma16(a[i], b[j], acc[i][j]);
  }
#pragma unroll
  for (int i = 0; i < 4; ++i)
#pragma unroll
    for (int j = 0; j < 4; ++j)
      store_tile(acc[i][j], mbase + 16 * i, nbase + 16 * j, lm, lh, Mout, bias, fout, hout, mode);
}

// ---------------- causal (dilated / strided) conv as implicit WMMA GEMM ----------------
// out[b, t, o] = act(bias[o] + sum_{k,i} w[k,o,i] * in[b, stride*t + k*dil - (K-1)*dil, i])
// grid.x = Mout/64, grid.y = Tout/128, grid.z = B
__global__ __launch_bounds__(256) void k_wmma_conv(const _Float16* __restrict__ in16,
                                                   const _Float16* __restrict__ w16,
                                                   const float* __restrict__ bias,
                                                   float* __restrict__ fout,
                                                   _Float16* __restrict__ hout,
                                                   int stride, int dil, int ksize,
                                                   int Cin, int Mout, int Tin, int Tout,
                                                   int mode) {
  const int lane = threadIdx.x & 31, wave = threadIdx.x >> 5;
  const int lm = lane & 15, lh = lane >> 4;
  const int mbase = blockIdx.x * 64 + (wave & 1) * 32;
  const int tbase = blockIdx.y * 128 + (wave >> 1) * 32;
  const int b = blockIdx.z;
  const int t0 = tbase + lm;
  const int t1 = t0 + 16;

  v8f c00 = {}, c01 = {}, c10 = {}, c11 = {};
  for (int kk = 0; kk < ksize; ++kk) {
    const int sh = (kk - (ksize - 1)) * dil;
    const int ts0 = stride * t0 + sh;
    const int ts1 = stride * t1 + sh;
    const _Float16* a0p = w16 + ((size_t)kk * Mout + mbase + lm) * Cin + lh * 16;
    const _Float16* a1p = a0p + (size_t)16 * Cin;
    const _Float16* b0p = in16 + ((size_t)b * Tin + ts0) * Cin + lh * 16;
    const _Float16* b1p = in16 + ((size_t)b * Tin + ts1) * Cin + lh * 16;
    for (int k0 = 0; k0 < Cin; k0 += 32) {
      FragH a0, a1, b0, b1;
      a0.h[0] = *(const v8h*)(a0p + k0);
      a0.h[1] = *(const v8h*)(a0p + k0 + 8);
      a1.h[0] = *(const v8h*)(a1p + k0);
      a1.h[1] = *(const v8h*)(a1p + k0 + 8);
      if (ts0 >= 0) b0.f = *(const v16h*)(b0p + k0); else frag_zero(b0);
      if (ts1 >= 0) b1.f = *(const v16h*)(b1p + k0); else frag_zero(b1);
      c00 = wmma16(a0, b0, c00);
      c01 = wmma16(a0, b1, c01);
      c10 = wmma16(a1, b0, c10);
      c11 = wmma16(a1, b1, c11);
    }
  }
  const size_t nb = (size_t)b * Tout + tbase;
  store_tile(c00, mbase,      nb,      lm, lh, Mout, bias, fout, hout, mode);
  store_tile(c01, mbase,      nb + 16, lm, lh, Mout, bias, fout, hout, mode);
  store_tile(c10, mbase + 16, nb,      lm, lh, Mout, bias, fout, hout, mode);
  store_tile(c11, mbase + 16, nb + 16, lm, lh, Mout, bias, fout, hout, mode);
}

// ---------------- LayerNorm over D=512 (one wave per row) ----------------
// mode 0: f16 out [row][512];  mode 1: f32 transposed out[(b*512+c)*Tloc + t]
__global__ __launch_bounds__(256) void k_layernorm(const float* __restrict__ in,
                                                   const float* __restrict__ g,
                                                   const float* __restrict__ bv,
                                                   _Float16* __restrict__ hout,
                                                   float* __restrict__ fout,
                                                   int Tloc, int mode) {
  const int lane = threadIdx.x & 31, wave = threadIdx.x >> 5;
  const int row = blockIdx.x * 8 + wave;
  const float* p = in + (size_t)row * 512;
  float x[16];
  float s = 0.f;
#pragma unroll
  for (int j = 0; j < 16; ++j) { x[j] = p[lane + j * 32]; s += x[j]; }
  s += __shfl_xor(s, 1, 32);  s += __shfl_xor(s, 2, 32);
  s += __shfl_xor(s, 4, 32);  s += __shfl_xor(s, 8, 32);
  s += __shfl_xor(s, 16, 32);
  float mean = s * (1.f / 512.f);
  float v = 0.f;
#pragma unroll
  for (int j = 0; j < 16; ++j) { float d = x[j] - mean; v += d * d; }
  v += __shfl_xor(v, 1, 32);  v += __shfl_xor(v, 2, 32);
  v += __shfl_xor(v, 4, 32);  v += __shfl_xor(v, 8, 32);
  v += __shfl_xor(v, 16, 32);
  float rstd = rsqrtf(v * (1.f / 512.f) + 1e-5f);
  if (mode == 0) {
    _Float16* ho = hout + (size_t)row * 512;
#pragma unroll
    for (int j = 0; j < 16; ++j) {
      int c = lane + j * 32;
      ho[c] = (_Float16)((x[j] - mean) * rstd * g[c] + bv[c]);
    }
  } else {
    int b = row / Tloc, t = row % Tloc;
#pragma unroll
    for (int j = 0; j < 16; ++j) {
      int c = lane + j * 32;
      fout[((size_t)(b * 512 + c)) * Tloc + t] = (x[j] - mean) * rstd * g[c] + bv[c];
    }
  }
}

// ---------------- sliding-window causal attention (flash, in registers) ----------------
// one wave = (b, h, 16-query tile). S^T = K.Q^T via WMMA; O^T = V^T.P^T via WMMA.
__global__ __launch_bounds__(256) void k_attn(const _Float16* __restrict__ qkv,
                                              const _Float16* __restrict__ vt,
                                              _Float16* __restrict__ ao) {
  const int lane = threadIdx.x & 31, wave = threadIdx.x >> 5;
  const int id = blockIdx.x * 8 + wave;        // 4096 tasks
  const int qt = id & 127;
  const int h = (id >> 7) & 7;
  const int b = id >> 10;
  const int q0 = qt * 16;
  const int lm = lane & 15, lh = lane >> 4;

  const _Float16* qbase = qkv + ((size_t)b * 2048) * 1536 + h * 64;
  const _Float16* kbase = qbase + 512;
  const _Float16* vbase = vt + ((size_t)(b * 8 + h) * 64) * 2048;

  FragH bq0, bq1;
  {
    const _Float16* qr = qbase + (size_t)(q0 + lm) * 1536 + lh * 16;
    bq0.f = *(const v16h*)(qr);
    bq1.f = *(const v16h*)(qr + 32);
  }

  v8f o0a = {}, o1a = {}, o2a = {}, o3a = {};
  float mrun = -1e30f, lrun = 0.f;
  const int q = q0 + lm;
  int klo = q0 - 127;
  klo = klo > 0 ? (klo & ~31) : 0;

  for (int ks = klo; ks <= q0 + 15; ks += 32) {
    v8f st0 = {}, st1 = {};
    {
      const _Float16* kr0 = kbase + (size_t)(ks + lm) * 1536 + lh * 16;
      const _Float16* kr1 = kr0 + (size_t)16 * 1536;
      FragH ak;
      ak.h[0] = *(const v8h*)(kr0);       ak.h[1] = *(const v8h*)(kr0 + 8);
      st0 = wmma16(ak, bq0, st0);
      ak.h[0] = *(const v8h*)(kr0 + 32);  ak.h[1] = *(const v8h*)(kr0 + 40);
      st0 = wmma16(ak, bq1, st0);
      ak.h[0] = *(const v8h*)(kr1);       ak.h[1] = *(const v8h*)(kr1 + 8);
      st1 = wmma16(ak, bq0, st1);
      ak.h[0] = *(const v8h*)(kr1 + 32);  ak.h[1] = *(const v8h*)(kr1 + 40);
      st1 = wmma16(ak, bq1, st1);
    }
    float s0[8], s1[8];
    float mc = -2e30f;
#pragma unroll
    for (int r = 0; r < 8; ++r) {
      int key0 = ks + r + 8 * lh;
      int key1 = key0 + 16;
      bool ok0 = (key0 <= q) && (key0 > q - 128);
      bool ok1 = (key1 <= q) && (key1 > q - 128);
      s0[r] = ok0 ? st0[r] * 0.125f : -2e30f;
      s1[r] = ok1 ? st1[r] * 0.125f : -2e30f;
      mc = fmaxf(mc, fmaxf(s0[r], s1[r]));
    }
    mc = fmaxf(mc, __shfl_xor(mc, 16, 32));
    float mnew = fmaxf(mrun, mc);
    float corr = expf(mrun - mnew);
    float rsum = 0.f;
#pragma unroll
    for (int r = 0; r < 8; ++r) {
      s0[r] = expf(s0[r] - mnew);
      s1[r] = expf(s1[r] - mnew);
      rsum += s0[r] + s1[r];
    }
    rsum += __shfl_xor(rsum, 16, 32);
    lrun = lrun * corr + rsum;
    mrun = mnew;
#pragma unroll
    for (int r = 0; r < 8; ++r) {
      o0a[r] *= corr; o1a[r] *= corr; o2a[r] *= corr; o3a[r] *= corr;
    }
    // repack P^T into B-fragment layout (xor-16 half exchange)
    FragH pb;
#pragma unroll
    for (int r = 0; r < 8; ++r) {
      float x0 = __shfl_xor(s0[r], 16, 32);
      float x1 = __shfl_xor(s1[r], 16, 32);
      pb.e[r]     = (_Float16)(lh ? x1 : s0[r]);
      pb.e[8 + r] = (_Float16)(lh ? s1[r] : x0);
    }
    const _Float16* vr = vbase + (size_t)lm * 2048 + ks + lh * 16;
    FragH av;
    av.h[0] = *(const v8h*)(vr);                  av.h[1] = *(const v8h*)(vr + 8);
    o0a = wmma16(av, pb, o0a);
    av.h[0] = *(const v8h*)(vr + 16 * 2048);      av.h[1] = *(const v8h*)(vr + 16 * 2048 + 8);
    o1a = wmma16(av, pb, o1a);
    av.h[0] = *(const v8h*)(vr + 32 * 2048);      av.h[1] = *(const v8h*)(vr + 32 * 2048 + 8);
    o2a = wmma16(av, pb, o2a);
    av.h[0] = *(const v8h*)(vr + 48 * 2048);      av.h[1] = *(const v8h*)(vr + 48 * 2048 + 8);
    o3a = wmma16(av, pb, o3a);
  }
  float linv = 1.f / lrun;
  _Float16* orow = ao + ((size_t)(b * 2048 + q0 + lm)) * 512 + h * 64;
#pragma unroll
  for (int r = 0; r < 8; ++r) {
    int dd = r + 8 * lh;
    orow[dd]      = (_Float16)(o0a[r] * linv);
    orow[16 + dd] = (_Float16)(o1a[r] * linv);
    orow[32 + dd] = (_Float16)(o2a[r] * linv);
    orow[48 + dd] = (_Float16)(o3a[r] * linv);
  }
}

// ---------------- host ----------------
extern "C" void kernel_launch(void* const* d_in, const int* in_sizes, int n_in,
                              void* d_out, int out_size, void* d_ws, size_t ws_size,
                              hipStream_t stream) {
  (void)in_sizes; (void)n_in; (void)out_size; (void)ws_size;
  const int B = 4, C = 128, T = 8192, D = 512, T2 = 2048, F = 2048;
  const int DILS[3] = {1, 3, 9};

  const float* x = (const float*)d_in[0];
  const float *cw1[3], *cb1[3], *cw2[3], *cb2[3];
  for (int r = 0; r < 3; ++r) {
    cw1[r] = (const float*)d_in[1 + 4 * r];
    cb1[r] = (const float*)d_in[2 + 4 * r];
    cw2[r] = (const float*)d_in[3 + 4 * r];
    cb2[r] = (const float*)d_in[4 + 4 * r];
  }
  const float* down_w = (const float*)d_in[13];
  const float* down_b = (const float*)d_in[14];
  const float *ln1g[4], *ln1b[4], *wqkv[4], *wout[4], *ln2g[4], *ln2b[4],
              *fw1[4], *fb1[4], *fw2[4], *fb2[4];
  for (int L = 0; L < 4; ++L) {
    int base = 15 + 10 * L;
    ln1g[L] = (const float*)d_in[base + 0];
    ln1b[L] = (const float*)d_in[base + 1];
    wqkv[L] = (const float*)d_in[base + 2];
    wout[L] = (const float*)d_in[base + 3];
    ln2g[L] = (const float*)d_in[base + 4];
    ln2b[L] = (const float*)d_in[base + 5];
    fw1[L]  = (const float*)d_in[base + 6];
    fb1[L]  = (const float*)d_in[base + 7];
    fw2[L]  = (const float*)d_in[base + 8];
    fb2[L]  = (const float*)d_in[base + 9];
  }
  const float* lnfg = (const float*)d_in[55];
  const float* lnfb = (const float*)d_in[56];

  // workspace arena
  char* W = (char*)d_ws;
  size_t off = 0;
  auto alloc = [&](size_t bytes) -> char* {
    char* p = W + off;
    off += bytes;
    off = (off + 255) & ~(size_t)255;
    return p;
  };
  _Float16 *wc1_16[3], *wc2_16[3];
  for (int r = 0; r < 3; ++r) {
    wc1_16[r] = (_Float16*)alloc((size_t)7 * C * C * 2);
    wc2_16[r] = (_Float16*)alloc((size_t)C * C * 2);
  }
  _Float16* wdn16 = (_Float16*)alloc((size_t)8 * D * C * 2);
  _Float16 *wqkv16[4], *wout16[4], *fw1_16[4], *fw2_16[4];
  for (int L = 0; L < 4; ++L) {
    wqkv16[L] = (_Float16*)alloc((size_t)3 * D * D * 2);
    wout16[L] = (_Float16*)alloc((size_t)D * D * 2);
    fw1_16[L] = (_Float16*)alloc((size_t)F * D * 2);
    fw2_16[L] = (_Float16*)alloc((size_t)D * F * 2);
  }
  const size_t NA = (size_t)B * T * C;   // == B*T2*D == 4194304
  float*    ACT  = (float*)alloc(NA * 4);
  float*    ACT2 = (float*)alloc(NA * 4);
  _Float16* H16  = (_Float16*)alloc(NA * 2);
  _Float16* Y16  = (_Float16*)alloc(NA * 2);
  _Float16* LN16 = (_Float16*)alloc(NA * 2);
  _Float16* VT16 = (_Float16*)alloc(NA * 2);
  _Float16* AO16 = (_Float16*)alloc(NA * 2);
  _Float16* QKV16 = (_Float16*)alloc((size_t)B * T2 * 3 * D * 2);
  _Float16* FFN16 = (_Float16*)alloc((size_t)B * T2 * F * 2);

  auto cdiv = [](size_t a, size_t b) { return (unsigned)((a + b - 1) / b); };

  // input transpose [B,C,T] -> [B,T,C]
  k_transpose_in<<<cdiv(NA, 256), 256, 0, stream>>>(x, ACT, C, T, (int)NA);

  // weight conversion / packing
  for (int r = 0; r < 3; ++r) {
    k_convert_w<<<cdiv((size_t)7 * C * C, 256), 256, 0, stream>>>(cw1[r], wc1_16[r], C, C, 7, 7 * C * C);
    k_convert_w<<<cdiv((size_t)C * C, 256), 256, 0, stream>>>(cw2[r], wc2_16[r], C, C, 1, C * C);
  }
  k_convert_w<<<cdiv((size_t)8 * D * C, 256), 256, 0, stream>>>(down_w, wdn16, D, C, 8, 8 * D * C);
  for (int L = 0; L < 4; ++L) {
    k_convert_w<<<cdiv((size_t)3 * D * D, 256), 256, 0, stream>>>(wqkv[L], wqkv16[L], 3 * D, D, 1, 3 * D * D);
    k_convert_w<<<cdiv((size_t)D * D, 256), 256, 0, stream>>>(wout[L], wout16[L], D, D, 1, D * D);
    k_convert_w<<<cdiv((size_t)F * D, 256), 256, 0, stream>>>(fw1[L], fw1_16[L], F, D, 1, F * D);
    k_convert_w<<<cdiv((size_t)D * F, 256), 256, 0, stream>>>(fw2[L], fw2_16[L], D, F, 1, D * F);
  }

  // residual dilated conv blocks
  for (int r = 0; r < 3; ++r) {
    k_elu16<<<cdiv(NA, 256), 256, 0, stream>>>(ACT, H16, (int)NA);
    k_wmma_conv<<<dim3(C / 64, T / 128, B), 256, 0, stream>>>(
        H16, wc1_16[r], cb1[r], nullptr, Y16, 1, DILS[r], 7, C, C, T, T, 4 /*ELU->f16*/);
    k_wmma_linear<<<dim3(C / 128, (B * T) / 256), 256, 0, stream>>>(
        Y16, wc2_16[r], cb2[r], ACT, nullptr, C, C, 2 /*+= f32*/);
  }

  // strided causal downsample -> [B, T2, D] f32 residual stream
  k_elu16<<<cdiv(NA, 256), 256, 0, stream>>>(ACT, H16, (int)NA);
  k_wmma_conv<<<dim3(D / 64, T2 / 128, B), 256, 0, stream>>>(
      H16, wdn16, down_b, ACT2, nullptr, 4, 1, 8, C, D, T, T2, 3 /*= f32*/);

  // transformer layers
  for (int L = 0; L < 4; ++L) {
    k_layernorm<<<(B * T2) / 8, 256, 0, stream>>>(ACT2, ln1g[L], ln1b[L], LN16, nullptr, T2, 0);
    k_wmma_linear<<<dim3((3 * D) / 128, (B * T2) / 256), 256, 0, stream>>>(
        LN16, wqkv16[L], nullptr, nullptr, QKV16, D, 3 * D, 0 /*f16*/);
    k_vtrans<<<cdiv(NA, 256), 256, 0, stream>>>(QKV16, VT16);
    k_attn<<<(B * 8 * (T2 / 16)) / 8, 256, 0, stream>>>(QKV16, VT16, AO16);
    k_wmma_linear<<<dim3(D / 128, (B * T2) / 256), 256, 0, stream>>>(
        AO16, wout16[L], nullptr, ACT2, nullptr, D, D, 2 /*+= f32*/);
    k_layernorm<<<(B * T2) / 8, 256, 0, stream>>>(ACT2, ln2g[L], ln2b[L], LN16, nullptr, T2, 0);
    k_wmma_linear<<<dim3(F / 128, (B * T2) / 256), 256, 0, stream>>>(
        LN16, fw1_16[L], fb1[L], nullptr, FFN16, D, F, 1 /*GELU->f16*/);
    k_wmma_linear<<<dim3(D / 128, (B * T2) / 256), 256, 0, stream>>>(
        FFN16, fw2_16[L], fb2[L], ACT2, nullptr, F, D, 2 /*+= f32*/);
  }

  // final LN, transposed store to d_out [B, 512, T2] f32
  k_layernorm<<<(B * T2) / 8, 256, 0, stream>>>(ACT2, lnfg, lnfb, nullptr, (float*)d_out, T2, 1);
}
